// CausalSelfAttention_73675868995713
// MI455X (gfx1250) — compile-verified, matrix operand order checked by reading, not run on hardware
//
#include <hip/hip_runtime.h>
#include <hip/hip_bf16.h>

typedef __attribute__((ext_vector_type(16))) __bf16 v16bf;
typedef __attribute__((ext_vector_type(8)))  __bf16 v8bf;
typedef __attribute__((ext_vector_type(8)))  float  v8f;
typedef unsigned int v4u __attribute__((ext_vector_type(4)));
typedef int          v8i __attribute__((ext_vector_type(8)));
typedef int          v4i __attribute__((ext_vector_type(4)));

#if defined(__HIP_DEVICE_COMPILE__) && __has_builtin(__builtin_amdgcn_tensor_load_to_lds)
#define USE_TDM 1
#else
#define USE_TDM 0
#endif

// ---------------------------------------------------------------------------
// WMMA fragment helper: A/B operand layout for v_wmma_f32_16x16x32_bf16.
// lane l (row = l&15), elements j<8 -> K = koff + j ; j>=8 -> K = 16+koff+(j-8)
// where koff = (l&16) ? 8 : 0.  Both 8-element runs are 16B-contiguous.
__device__ __forceinline__ v16bf load_frag16(const __bf16* p) {
  v8bf lo = *(const v8bf*)p;
  v8bf hi = *(const v8bf*)(p + 16);
  v16bf o;
#pragma unroll
  for (int i = 0; i < 8; ++i) { o[i] = lo[i]; o[i + 8] = hi[i]; }
  return o;
}

__device__ __forceinline__ v8f wmma_bf16(v16bf a, v16bf b, v8f c) {
  return __builtin_amdgcn_wmma_f32_16x16x32_bf16(false, a, false, b,
                                                 (short)0, c, false, false);
}

#if USE_TDM
// Issue one TDM 2D tile load: 128 rows x 32 bf16, global row stride K elements,
// LDS destination padded to 40-element (80B) rows via D# pad fields.
__device__ __forceinline__ void tdm_load_tile(const __bf16* gsrc,
                                              unsigned lds_addr, int K) {
  const unsigned long long ga = (unsigned long long)gsrc;
  v4u g0;
  g0[0] = 1u;                                   // count=1, no gather
  g0[1] = lds_addr;                             // LDS byte address
  g0[2] = (unsigned)ga;                         // global_addr[31:0]
  g0[3] = (unsigned)((ga >> 32) & 0x01FFFFFFu)  // global_addr[56:32]
          | (2u << 30);                         // type=2 ("image")
  v8i g1;
  // data_size=1(2B) | pad_enable | pad_interval=3(64B) | pad_amount=3(16B)
  g1[0] = (int)((1u << 16) | (1u << 20) | (3u << 22) | (3u << 25));
  g1[1] = (int)(((unsigned)K & 0xFFFFu) << 16);          // tensor_dim0 lo
  g1[2] = (int)((((unsigned)K >> 16) & 0xFFFFu)          // tensor_dim0 hi
                | (128u << 16));                          // tensor_dim1 = 128
  g1[3] = (int)(32u << 16);                               // tile_dim0 = 32
  g1[4] = 128;                                            // tile_dim1 = 128
  g1[5] = K;                                              // dim0_stride lo
  g1[6] = 0;
  g1[7] = 0;
  v4i z4 = {0, 0, 0, 0};
  v8i z8 = {0, 0, 0, 0, 0, 0, 0, 0};
  // 6-arg toolchain signature: (g0 v4u, g1 v8i, v4i, v4i, v8i, i32 cpol)
  __builtin_amdgcn_tensor_load_to_lds(g0, g1, z4, z4, z8, 0);
}
#endif

// ---------------------------------------------------------------------------
// fp32 -> bf16 conversion (grid-stride)
// ---------------------------------------------------------------------------
__global__ void f32_to_bf16_kernel(const float* __restrict__ in,
                                   __bf16* __restrict__ out, size_t n) {
  size_t i = (size_t)blockIdx.x * blockDim.x + threadIdx.x;
  size_t stride = (size_t)gridDim.x * blockDim.x;
  for (; i < n; i += stride) out[i] = (__bf16)in[i];
}

// ---------------------------------------------------------------------------
// Tiled WMMA GEMM:  C[M,N] = A[M,K] * W[N,K]^T   (A,W bf16 row-major)
// Tiles staged into double-buffered LDS by the Tensor Data Mover (wave 0),
// overlapped with WMMA compute on the other buffer.
// MODE 0: write f32 row-major to outF
// MODE 1: scatter q/k into [B,H,T,D] bf16, v transposed into [B,H,D,T] bf16
// ---------------------------------------------------------------------------
template <int MODE>
__global__ __launch_bounds__(256) void gemm_bf16_kernel(
    const __bf16* __restrict__ A, const __bf16* __restrict__ W,
    float* __restrict__ outF,
    __bf16* __restrict__ qb, __bf16* __restrict__ kb, __bf16* __restrict__ vtb,
    int M, int N, int K) {
  constexpr int BM = 128, BN = 128, BK = 32, LDSS = 40;  // padded row stride
  __shared__ __align__(16) __bf16 As[2][BM * LDSS];
  __shared__ __align__(16) __bf16 Bs[2][BN * LDSS];

  const int tid  = threadIdx.x;
  const int lane = tid & 31;
  const int wave = tid >> 5;
  const int m0 = blockIdx.y * BM;
  const int n0 = blockIdx.x * BN;
  const int wm = (wave & 3) * 32;   // 4 wave-rows (M)
  const int wn = (wave >> 2) * 64;  // 2 wave-cols (N)

  const int koff = (lane & 16) ? 8 : 0;
  const int lrow = lane & 15;

  v8f acc[2][4] = {};

#if USE_TDM
  const unsigned ldsA0 = (unsigned)(unsigned long long)(void*)&As[0][0];
  const unsigned ldsA1 = (unsigned)(unsigned long long)(void*)&As[1][0];
  const unsigned ldsB0 = (unsigned)(unsigned long long)(void*)&Bs[0][0];
  const unsigned ldsB1 = (unsigned)(unsigned long long)(void*)&Bs[1][0];
  if (wave == 0) {
    tdm_load_tile(A + (size_t)m0 * K, ldsA0, K);
    tdm_load_tile(W + (size_t)n0 * K, ldsB0, K);
    __builtin_amdgcn_s_wait_tensorcnt(0);
  }
  __syncthreads();

  for (int k0 = 0; k0 < K; k0 += BK) {
    const int cur = (k0 >> 5) & 1;
    const bool more = (k0 + BK) < K;
    if (wave == 0 && more) {  // prefetch next tile into the other buffer
      tdm_load_tile(A + (size_t)m0 * K + k0 + BK, cur ? ldsA0 : ldsA1, K);
      tdm_load_tile(W + (size_t)n0 * K + k0 + BK, cur ? ldsB0 : ldsB1, K);
    }
    v16bf af[2], bfr[4];
#pragma unroll
    for (int i = 0; i < 2; ++i)
      af[i] = load_frag16(&As[cur][0] + (wm + i * 16 + lrow) * LDSS + koff);
#pragma unroll
    for (int j = 0; j < 4; ++j)
      bfr[j] = load_frag16(&Bs[cur][0] + (wn + j * 16 + lrow) * LDSS + koff);
#pragma unroll
    for (int i = 0; i < 2; ++i)
#pragma unroll
      for (int j = 0; j < 4; ++j)
        acc[i][j] = wmma_bf16(af[i], bfr[j], acc[i][j]);
    asm volatile("s_wait_dscnt 0" ::: "memory");
    if (wave == 0 && more) __builtin_amdgcn_s_wait_tensorcnt(0);
    __syncthreads();
  }
#else
  // Fallback: manual staging (each thread moves 16 bf16 = two 16B chunks)
  const int lr = tid >> 1;
  const int lh = (tid & 1) * 16;
  for (int k0 = 0; k0 < K; k0 += BK) {
    __syncthreads();
    {
      const __bf16* sa = A + (size_t)(m0 + lr) * K + k0 + lh;
      const __bf16* sw = W + (size_t)(n0 + lr) * K + k0 + lh;
      *(uint4*)(&As[0][0] + lr * LDSS + lh)     = *(const uint4*)sa;
      *(uint4*)(&As[0][0] + lr * LDSS + lh + 8) = *(const uint4*)(sa + 8);
      *(uint4*)(&Bs[0][0] + lr * LDSS + lh)     = *(const uint4*)sw;
      *(uint4*)(&Bs[0][0] + lr * LDSS + lh + 8) = *(const uint4*)(sw + 8);
      if (k0 + BK < K) {
        __builtin_prefetch(sa + BK, 0, 1);
        __builtin_prefetch(sw + BK, 0, 1);
      }
    }
    __syncthreads();
    v16bf af[2], bfr[4];
#pragma unroll
    for (int i = 0; i < 2; ++i)
      af[i] = load_frag16(&As[0][0] + (wm + i * 16 + lrow) * LDSS + koff);
#pragma unroll
    for (int j = 0; j < 4; ++j)
      bfr[j] = load_frag16(&Bs[0][0] + (wn + j * 16 + lrow) * LDSS + koff);
#pragma unroll
    for (int i = 0; i < 2; ++i)
#pragma unroll
      for (int j = 0; j < 4; ++j)
        acc[i][j] = wmma_bf16(af[i], bfr[j], acc[i][j]);
  }
#endif

  // epilogue: C/D layout -> lane l holds col n0+wn+j*16+(l&15),
  // vgpr r holds row m0+wm+i*16+r+((l&16)?8:0)
  const int rbase = (lane & 16) ? 8 : 0;
#pragma unroll
  for (int i = 0; i < 2; ++i) {
#pragma unroll
    for (int j = 0; j < 4; ++j) {
      const int colg = n0 + wn + j * 16 + lrow;
#pragma unroll
      for (int r = 0; r < 8; ++r) {
        const int rowg = m0 + wm + i * 16 + rbase + r;
        const float v = acc[i][j][r];
        if constexpr (MODE == 0) {
          outF[(size_t)rowg * N + colg] = v;
        } else {
          const int part = colg >> 11;      // 0=q 1=k 2=v
          const int c = colg & 2047;
          const int h = c >> 7, d = c & 127;
          const int b = rowg >> 11, t = rowg & 2047;
          const __bf16 bv = (__bf16)v;
          const size_t bh = (size_t)b * 16 + h;
          if (part == 0)      qb [(bh * 2048 + t) * 128 + d] = bv;
          else if (part == 1) kb [(bh * 2048 + t) * 128 + d] = bv;
          else                vtb[(bh * 128 + d) * 2048 + t] = bv;  // V^T
        }
      }
    }
  }
}

// ---------------------------------------------------------------------------
// RoPE in-place on q,k bf16 [B,H,T,D]; one thread per (b,h,t,d<D/2)
// ---------------------------------------------------------------------------
__global__ __launch_bounds__(256) void rope_kernel(__bf16* __restrict__ q,
                                                   __bf16* __restrict__ k) {
  const size_t total = (size_t)4 * 16 * 2048 * 64;
  size_t idx = (size_t)blockIdx.x * blockDim.x + threadIdx.x;
  if (idx >= total) return;
  const int d = (int)(idx & 63);
  const size_t rest = idx >> 6;
  const int t = (int)(rest & 2047);
  const size_t bh = rest >> 11;
  const size_t base = (bh * 2048 + t) * 128;
  const float freq = (float)t * __powf(10000.0f, -(float)(2 * d) / 128.0f);
  float sn, cs;
  __sincosf(freq, &sn, &cs);
  {
    const float x1 = (float)q[base + d], x2 = (float)q[base + d + 64];
    q[base + d]      = (__bf16)(x1 * cs - x2 * sn);
    q[base + d + 64] = (__bf16)(x2 * cs + x1 * sn);
  }
  {
    const float x1 = (float)k[base + d], x2 = (float)k[base + d + 64];
    k[base + d]      = (__bf16)(x1 * cs - x2 * sn);
    k[base + d + 64] = (__bf16)(x2 * cs + x1 * sn);
  }
}

// ---------------------------------------------------------------------------
// Flash attention: one wave per (b,h, 16-query tile), 32 keys / iteration.
// q,k: [B,H,T,D] bf16; vt: [B,H,D,T] bf16; out: [B,T,C] bf16
// ---------------------------------------------------------------------------
__global__ __launch_bounds__(128) void flash_attn_kernel(
    const __bf16* __restrict__ q, const __bf16* __restrict__ k,
    const __bf16* __restrict__ vt, __bf16* __restrict__ out) {
  __shared__ __align__(16) __bf16 pbuf[4][16 * 32];  // per-wave P staging
  const int lane  = threadIdx.x & 31;
  const int wv    = threadIdx.x >> 5;
  const int gwave = blockIdx.x * 4 + wv;
  const int qt = gwave & 127;   // T/16 tiles
  const int bh = gwave >> 7;    // b*16+h
  const int qbase = qt * 16;
  const int lrow  = lane & 15;
  const int koff  = (lane & 16) ? 8 : 0;
  const int rbase = (lane & 16) ? 8 : 0;

  const __bf16* qp = q  + (size_t)bh * 2048 * 128;
  const __bf16* kp = k  + (size_t)bh * 2048 * 128;
  const __bf16* vp = vt + (size_t)bh * 128 * 2048;
  __bf16* pw = pbuf[wv];

  v16bf qf[4];
#pragma unroll
  for (int c = 0; c < 4; ++c)
    qf[c] = load_frag16(qp + (size_t)(qbase + lrow) * 128 + c * 32 + koff);

  v8f O[8] = {};
  float mrow[8], lsum[8];
#pragma unroll
  for (int r = 0; r < 8; ++r) { mrow[r] = -1e30f; lsum[r] = 0.0f; }

  const float sc = 0.08838834764831845f;  // 1/sqrt(128)

  for (int kb = 0; kb < qbase + 16; kb += 32) {
    v8f S0 = {}, S1 = {};
#pragma unroll
    for (int c = 0; c < 4; ++c) {
      v16bf b0 = load_frag16(kp + (size_t)(kb + lrow) * 128 + c * 32 + koff);
      S0 = wmma_bf16(qf[c], b0, S0);
    }
#pragma unroll
    for (int c = 0; c < 4; ++c) {
      v16bf b1 = load_frag16(kp + (size_t)(kb + 16 + lrow) * 128 + c * 32 + koff);
      S1 = wmma_bf16(qf[c], b1, S1);
    }
    const int col0 = kb + lrow;
#pragma unroll
    for (int r = 0; r < 8; ++r) {
      const int qrow = qbase + rbase + r;
      float s0 = (col0      <= qrow) ? S0[r] * sc : -1e30f;
      float s1 = (col0 + 16 <= qrow) ? S1[r] * sc : -1e30f;
      float mx = fmaxf(s0, s1);
#pragma unroll
      for (int off = 1; off < 16; off <<= 1)
        mx = fmaxf(mx, __shfl_xor(mx, off, 32));   // stays within 16-lane half
      const float nm = fmaxf(mrow[r], mx);
      const float p0 = __expf(s0 - nm);
      const float p1 = __expf(s1 - nm);
      float ps = p0 + p1;
#pragma unroll
      for (int off = 1; off < 16; off <<= 1) ps += __shfl_xor(ps, off, 32);
      const float rescale = __expf(mrow[r] - nm);
      lsum[r] = lsum[r] * rescale + ps;
      mrow[r] = nm;
#pragma unroll
      for (int nt = 0; nt < 8; ++nt) O[nt][r] *= rescale;
      pw[(rbase + r) * 32 + lrow]      = (__bf16)p0;  // C-layout -> row-major LDS
      pw[(rbase + r) * 32 + 16 + lrow] = (__bf16)p1;
    }
    asm volatile("s_wait_dscnt 0" ::: "memory");      // LDS RAW fence (same wave)
    v16bf pf = load_frag16(pw + lrow * 32 + koff);    // P as A-operand (16x32)
#pragma unroll
    for (int nt = 0; nt < 8; ++nt) {
      // B-operand: V^T row = dcol, contiguous over keys
      v16bf vf = load_frag16(vp + (size_t)(nt * 16 + lrow) * 2048 + kb + koff);
      O[nt] = wmma_bf16(pf, vf, O[nt]);
    }
  }

  const int b = bh >> 4, h = bh & 15;
#pragma unroll
  for (int r = 0; r < 8; ++r) {
    const float inv = 1.0f / lsum[r];
    const int qrow = qbase + rbase + r;
#pragma unroll
    for (int nt = 0; nt < 8; ++nt)
      out[((size_t)(b * 2048 + qrow)) * 2048 + h * 128 + nt * 16 + lrow] =
          (__bf16)(O[nt][r] * inv);
  }
}

// ---------------------------------------------------------------------------
// Host launcher
// ---------------------------------------------------------------------------
extern "C" void kernel_launch(void* const* d_in, const int* in_sizes, int n_in,
                              void* d_out, int out_size, void* d_ws,
                              size_t ws_size, hipStream_t stream) {
  (void)in_sizes; (void)n_in; (void)out_size; (void)ws_size;
  const float* x     = (const float*)d_in[0];
  const float* Wqkv  = (const float*)d_in[1];
  const float* Wproj = (const float*)d_in[2];
  float* out = (float*)d_out;

  constexpr size_t M = 8192, K = 2048, N1 = 6144, N2 = 2048;

  char* ws = (char*)d_ws;
  __bf16* xb     = (__bf16*)ws; ws += M * K * 2;    // 32 MB
  __bf16* wqkvb  = (__bf16*)ws; ws += N1 * K * 2;   // 24 MB
  __bf16* wprojb = (__bf16*)ws; ws += N2 * K * 2;   //  8 MB
  __bf16* qb     = (__bf16*)ws; ws += M * K * 2;    // 32 MB [B,H,T,D]
  __bf16* kb     = (__bf16*)ws; ws += M * K * 2;    // 32 MB [B,H,T,D]
  __bf16* vtb    = (__bf16*)ws; ws += M * K * 2;    // 32 MB [B,H,D,T]
  __bf16* attn   = (__bf16*)ws; ws += M * K * 2;    // 32 MB [B,T,C]

  f32_to_bf16_kernel<<<4096, 256, 0, stream>>>(x, xb, M * K);
  f32_to_bf16_kernel<<<4096, 256, 0, stream>>>(Wqkv, wqkvb, N1 * K);
  f32_to_bf16_kernel<<<2048, 256, 0, stream>>>(Wproj, wprojb, N2 * K);

  dim3 g1(N1 / 128, M / 128);
  gemm_bf16_kernel<1><<<g1, 256, 0, stream>>>(xb, wqkvb, nullptr, qb, kb, vtb,
                                              (int)M, (int)N1, (int)K);

  rope_kernel<<<32768, 256, 0, stream>>>(qb, kb);

  flash_attn_kernel<<<8192 / 4, 128, 0, stream>>>(qb, kb, vtb, attn);

  dim3 g2(N2 / 128, M / 128);
  gemm_bf16_kernel<0><<<g2, 256, 0, stream>>>(attn, wprojb, out, nullptr,
                                              nullptr, nullptr, (int)M, (int)N2,
                                              (int)K);
}
